// OGSympowAttention_83777632075958
// MI455X (gfx1250) — matrix operand order repaired and checked by reading, not compiled, not run on hardware
//
#include <hip/hip_runtime.h>
#include <hip/hip_bf16.h>
#include <stdint.h>

// ---------------- problem constants (from reference) ----------------
#define TT      2048
#define BB      2
#define HH      8
#define DD      64
#define HD      512      // H*D
#define NQ      1544     // 3*HD + H
#define NQP     1664     // NQ padded to 13*128 (GEMM tile grid)
#define CC      512
#define SCALE   0.125f   // 1/sqrt(64)
#define GATE_BIAS 6.906768f
#define TWO_PI  6.28318530717958647692f
#define LOG2_T  11.0f    // log2(2048)

typedef __bf16 bf16_t;
typedef __attribute__((ext_vector_type(16))) __bf16 v16bf;
typedef __attribute__((ext_vector_type(8)))  __bf16 v8bf;
typedef __attribute__((ext_vector_type(8)))  float  v8f;

// build a v16bf from two contiguous 16B halves without per-element moves
typedef union { v16bf v; v8bf h[2]; } v16u;

static __device__ __forceinline__ bf16_t to_bf16(float f) {
    unsigned int u = __float_as_uint(f);
    unsigned int r = u + 0x7FFFu + ((u >> 16) & 1u);   // round-to-nearest-even
    unsigned short h = (unsigned short)(r >> 16);
    bf16_t b;
    __builtin_memcpy(&b, &h, sizeof(b));
    return b;
}

// ---------------- kernel 1a: f32 -> bf16 (plain) ----------------
__global__ void cvt_bf16_kernel(const float* __restrict__ in, bf16_t* __restrict__ out, int n) {
    int i = blockIdx.x * blockDim.x + threadIdx.x;
    if (i < n) out[i] = to_bf16(in[i]);
}

// ---------------- kernel 1b: f32 -> bf16, transpose + zero-pad ----------------
// in:  [K][N] row-major;  out: [Npad][K] row-major (rows n >= N are zero)
__global__ void cvt_tr_bf16_kernel(const float* __restrict__ in, bf16_t* __restrict__ out,
                                   int K, int N, int Npad) {
    int idx = blockIdx.x * blockDim.x + threadIdx.x;
    if (idx < Npad * K) {
        int n = idx / K, k = idx - n * K;
        out[idx] = (n < N) ? to_bf16(in[(size_t)k * N + n]) : to_bf16(0.0f);
    }
}

// ---------------- kernel 2/5: bf16 GEMM, f32 out, no LDS ----------------
// C[M][ldc] = A[M][K] @ WT^T  where WT is [ldc][K] (B pre-transposed, n-major, zero padded).
// All fragment loads are contiguous global b128s; L2 (192MB) provides the reuse.
// Block = 256 thr (8 waves); block tile 64x128; wave tile 32x32 (2x2 WMMA).
__global__ __launch_bounds__(256)
void gemm_bf16_kernel(const bf16_t* __restrict__ A, const bf16_t* __restrict__ WT,
                      float* __restrict__ C, int K, int ldc) {
    const int lane = threadIdx.x & 31;
    const int wave = threadIdx.x >> 5;
    const int hi   = (lane >> 4) & 1;
    const int l15  = lane & 15;
    const int m0   = blockIdx.y * 64 + (wave & 1) * 32;
    const int n0   = blockIdx.x * 128 + (wave >> 1) * 32;

    const bf16_t* arow[2];
    const bf16_t* brow[2];
    arow[0] = A  + (size_t)(m0 + l15)      * K + hi * 8;   // A frag: K {0..7,16..23} / {8..15,24..31}
    arow[1] = A  + (size_t)(m0 + 16 + l15) * K + hi * 8;
    brow[0] = WT + (size_t)(n0 + l15)      * K + hi * 8;   // B frag: K {0..7,16..23} / {8..15,24..31}
    brow[1] = WT + (size_t)(n0 + 16 + l15) * K + hi * 8;

    v8f acc[2][2] = {};

    for (int k0 = 0; k0 < K; k0 += 32) {
        v16u a0, a1, b0, b1;
        a0.h[0] = *(const v8bf*)(arow[0] + k0);
        a0.h[1] = *(const v8bf*)(arow[0] + k0 + 16);
        a1.h[0] = *(const v8bf*)(arow[1] + k0);
        a1.h[1] = *(const v8bf*)(arow[1] + k0 + 16);
        b0.h[0] = *(const v8bf*)(brow[0] + k0);
        b0.h[1] = *(const v8bf*)(brow[0] + k0 + 16);
        b1.h[0] = *(const v8bf*)(brow[1] + k0);
        b1.h[1] = *(const v8bf*)(brow[1] + k0 + 16);
        acc[0][0] = __builtin_amdgcn_wmma_f32_16x16x32_bf16(false, a0.v, false, b0.v, (short)0, acc[0][0], false, false);
        acc[0][1] = __builtin_amdgcn_wmma_f32_16x16x32_bf16(false, a0.v, false, b1.v, (short)0, acc[0][1], false, false);
        acc[1][0] = __builtin_amdgcn_wmma_f32_16x16x32_bf16(false, a1.v, false, b0.v, (short)0, acc[1][0], false, false);
        acc[1][1] = __builtin_amdgcn_wmma_f32_16x16x32_bf16(false, a1.v, false, b1.v, (short)0, acc[1][1], false, false);
    }

    // store (C layout: lane holds column n=l15; VGPR i -> row i + 8*hi); ldc is padded -> no guard
    #pragma unroll
    for (int im = 0; im < 2; ++im)
        #pragma unroll
        for (int in = 0; in < 2; ++in) {
            int n = n0 + in * 16 + l15;
            #pragma unroll
            for (int i = 0; i < 8; ++i) {
                int m = m0 + im * 16 + i + hi * 8;
                C[(size_t)m * ldc + n] = acc[im][in][i];
            }
        }
}

// ---------------- kernel 3: gate cumsum + RoPE + repack ----------------
// one block per (b,h); q,k -> bf16 [bh][t][d] (roped); v -> bf16 [bh][d][t]; L f32 [bh][t]
__global__ __launch_bounds__(256)
void post_kernel(const float* __restrict__ qkvg,
                 bf16_t* __restrict__ qo, bf16_t* __restrict__ ko,
                 bf16_t* __restrict__ vt, float* __restrict__ L) {
    const int bh = blockIdx.x;
    const int b  = bh >> 3, h = bh & 7;
    const int tid = threadIdx.x;
    __shared__ float ssum[256];

    // --- inclusive cumsum of log_sigmoid gate over T ---
    float local[8];
    float run = 0.0f;
    const int tb = tid * 8;
    #pragma unroll
    for (int r = 0; r < 8; ++r) {
        int t = tb + r;
        float z  = GATE_BIAS + qkvg[(size_t)(b * TT + t) * NQP + 3 * HD + h];
        float lg = -log1pf(__expf(-z));
        run += lg;
        local[r] = run;
    }
    ssum[tid] = run;
    __syncthreads();
    for (int off = 1; off < 256; off <<= 1) {
        float v = (tid >= off) ? ssum[tid - off] : 0.0f;
        __syncthreads();
        ssum[tid] += v;
        __syncthreads();
    }
    float excl = ssum[tid] - run;
    #pragma unroll
    for (int r = 0; r < 8; ++r) L[(size_t)bh * TT + tb + r] = excl + local[r];

    // --- RoPE q,k ---
    for (int idx = tid; idx < TT * (DD / 2); idx += 256) {
        int t = idx >> 5, j = idx & 31;                      // pair index j in 0..31
        float div = TWO_PI * exp2f(-(float)j * (LOG2_T / 32.0f));  // 2pi / T^(j/32)
        float ang = (float)t * div;
        float s, c;
        __sincosf(ang, &s, &c);
        size_t base = (size_t)(b * TT + t) * NQP;
        size_t o    = ((size_t)bh * TT + t) * DD + 2 * j;
        float x1 = qkvg[base + h * DD + 2 * j];
        float x2 = qkvg[base + h * DD + 2 * j + 1];
        qo[o]     = to_bf16(x1 * c - x2 * s);
        qo[o + 1] = to_bf16(x2 * c + x1 * s);
        x1 = qkvg[base + HD + h * DD + 2 * j];
        x2 = qkvg[base + HD + h * DD + 2 * j + 1];
        ko[o]     = to_bf16(x1 * c - x2 * s);
        ko[o + 1] = to_bf16(x2 * c + x1 * s);
    }
    // --- V transposed [d][t] for B-fragment-friendly loads ---
    for (int idx = tid; idx < TT * DD; idx += 256) {
        int t = idx >> 6, d = idx & 63;
        vt[(size_t)bh * DD * TT + (size_t)d * TT + t] =
            to_bf16(qkvg[(size_t)(b * TT + t) * NQP + 2 * HD + h * DD + d]);
    }
}

// ---------------- kernel 4: causal gated sympow attention ----------------
// grid = B*H*(T/128); 8 waves/block, 16 query rows per wave.
__global__ __launch_bounds__(256)
void attn_kernel(const bf16_t* __restrict__ q, const bf16_t* __restrict__ k,
                 const bf16_t* __restrict__ vt, const float* __restrict__ L,
                 bf16_t* __restrict__ y) {
    const int blk    = blockIdx.x;
    const int tchunk = blk & 15;          // T/128 == 16
    const int bh     = blk >> 4;
    const int b      = bh >> 3, h = bh & 7;
    const int lane   = threadIdx.x & 31;
    const int wave   = threadIdx.x >> 5;
    const int hi     = (lane >> 4) & 1;
    const int l15    = lane & 15;
    const int t0     = tchunk * 128 + wave * 16;

    const bf16_t* qh = q  + (size_t)bh * TT * DD;
    const bf16_t* kh = k  + (size_t)bh * TT * DD;
    const bf16_t* vh = vt + (size_t)bh * DD * TT;
    const float*  Lh = L  + (size_t)bh * TT;

    __shared__ __align__(16) bf16_t sS[8][16][32];   // per-wave 16x32 score tile

    // Q A-fragments for d-blocks {0..31},{32..63}
    v16u aq[2];
    {
        const bf16_t* qrow = qh + (size_t)(t0 + l15) * DD;
        #pragma unroll
        for (int db = 0; db < 2; ++db) {
            int off = db * 32 + hi * 16;
            aq[db].h[0] = *(const v8bf*)(qrow + off);
            aq[db].h[1] = *(const v8bf*)(qrow + off + 8);
        }
    }
    float Lt[8];
    #pragma unroll
    for (int i = 0; i < 8; ++i) Lt[i] = Lh[t0 + i + hi * 8];

    float dsum[8] = {0, 0, 0, 0, 0, 0, 0, 0};
    v8f Y[4] = {};

    const int smax = t0 + 16;
    for (int s2 = 0; s2 < smax; s2 += 32) {
        // two 16-wide S tiles -> gated scores -> LDS (bf16)
        #pragma unroll
        for (int u = 0; u < 2; ++u) {
            int ss = s2 + u * 16;
            if (ss < smax) {
                const bf16_t* krow = kh + (size_t)(ss + l15) * DD;
                v16bf bk0 = *(const v16bf*)(krow + hi * 16);
                v16bf bk1 = *(const v16bf*)(krow + 32 + hi * 16);
                v8f c = {};
                c = __builtin_amdgcn_wmma_f32_16x16x32_bf16(false, aq[0].v, false, bk0, (short)0, c, false, false);
                c = __builtin_amdgcn_wmma_f32_16x16x32_bf16(false, aq[1].v, false, bk1, (short)0, c, false, false);
                int   s  = ss + l15;
                float Ls = Lh[s];
                #pragma unroll
                for (int i = 0; i < 8; ++i) {
                    int t = t0 + i + hi * 8;
                    float f   = c[i] * SCALE;
                    float val = (s <= t) ? f * f * __expf(Lt[i] - Ls) : 0.0f;
                    dsum[i] += val;
                    sS[wave][i + hi * 8][u * 16 + l15] = to_bf16(val);
                }
            } else {
                #pragma unroll
                for (int i = 0; i < 8; ++i)
                    sS[wave][i + hi * 8][u * 16 + l15] = to_bf16(0.0f);
            }
        }
        __builtin_amdgcn_wave_barrier();   // keep LDS RAW order (same-wave DS ops are in-order)

        // re-layout scores C->A fragment via LDS
        v16u aS;
        {
            int off = hi * 16;
            aS.h[0] = *(const v8bf*)&sS[wave][l15][off];
            aS.h[1] = *(const v8bf*)&sS[wave][l15][8 + off];
        }
        // Y += S @ V   (V stored [d][t] -> contiguous B fragments)
        #pragma unroll
        for (int j = 0; j < 4; ++j) {
            v16bf bv = *(const v16bf*)(vh + (size_t)(j * 16 + l15) * TT + s2 + hi * 16);
            Y[j] = __builtin_amdgcn_wmma_f32_16x16x32_bf16(false, aS.v, false, bv, (short)0, Y[j], false, false);
        }
    }

    // row-sum of denominator across 16-lane halves (wave32)
    #pragma unroll
    for (int i = 0; i < 8; ++i) {
        float v = dsum[i];
        v += __shfl_xor(v, 1);
        v += __shfl_xor(v, 2);
        v += __shfl_xor(v, 4);
        v += __shfl_xor(v, 8);
        dsum[i] = fmaxf(v, 1e-6f);
    }
    // store y (bf16, [b*T+t][h*64+d]) for the final GEMM
    #pragma unroll
    for (int j = 0; j < 4; ++j)
        #pragma unroll
        for (int i = 0; i < 8; ++i) {
            int t   = t0 + i + hi * 8;
            int col = h * DD + j * 16 + l15;
            y[(size_t)(b * TT + t) * HD + col] = to_bf16(Y[j][i] / dsum[i]);
        }
}

// ---------------- host launcher ----------------
extern "C" void kernel_launch(void* const* d_in, const int* in_sizes, int n_in,
                              void* d_out, int out_size, void* d_ws, size_t ws_size,
                              hipStream_t stream) {
    (void)in_sizes; (void)n_in; (void)out_size; (void)ws_size;
    const float* x      = (const float*)d_in[0];
    const float* w_attn = (const float*)d_in[1];
    const float* w_proj = (const float*)d_in[2];
    float* out = (float*)d_out;

    const size_t M  = (size_t)BB * TT;         // 4096
    const size_t NX = M * CC;                  // x elems

    char* ws = (char*)d_ws;
    size_t off = 0;
    auto carve = [&](size_t bytes) -> void* {
        void* p = ws + off;
        off = (off + bytes + 255) & ~(size_t)255;
        return p;
    };
    bf16_t* x_bf   = (bf16_t*)carve(NX * sizeof(bf16_t));
    bf16_t* wa_t   = (bf16_t*)carve((size_t)NQP * CC * sizeof(bf16_t));  // w_attn^T padded [1664][512]
    bf16_t* wp_t   = (bf16_t*)carve((size_t)CC  * HD * sizeof(bf16_t));  // w_proj^T        [512][512]
    float*  qkvg   = (float*) carve(M * NQP * sizeof(float));            // padded columns
    bf16_t* q_bf   = (bf16_t*)carve((size_t)BB * HH * TT * DD * sizeof(bf16_t));
    bf16_t* k_bf   = (bf16_t*)carve((size_t)BB * HH * TT * DD * sizeof(bf16_t));
    bf16_t* v_bf   = (bf16_t*)carve((size_t)BB * HH * DD * TT * sizeof(bf16_t));
    float*  Lbuf   = (float*) carve((size_t)BB * HH * TT * sizeof(float));
    bf16_t* y_bf   = (bf16_t*)carve(M * HD * sizeof(bf16_t));

    // 1) convert x; transpose+pad weights (B matrices become [n][k], K-contiguous)
    cvt_bf16_kernel<<<(int)((NX + 255) / 256), 256, 0, stream>>>(x, x_bf, (int)NX);
    cvt_tr_bf16_kernel<<<(NQP * CC + 255) / 256, 256, 0, stream>>>(w_attn, wa_t, CC, NQ, NQP);
    cvt_tr_bf16_kernel<<<(CC * HD + 255) / 256, 256, 0, stream>>>(w_proj, wp_t, HD, CC, CC);

    // 2) qkvg = x @ w_attn    (M=4096, Npad=1664, K=512)
    gemm_bf16_kernel<<<dim3(NQP / 128, M / 64), 256, 0, stream>>>(x_bf, wa_t, qkvg, CC, NQP);

    // 3) gate cumsum + RoPE + repack
    post_kernel<<<BB * HH, 256, 0, stream>>>(qkvg, q_bf, k_bf, v_bf, Lbuf);

    // 4) attention
    attn_kernel<<<BB * HH * (TT / 128), 256, 0, stream>>>(q_bf, k_bf, v_bf, Lbuf, y_bf);

    // 5) out = y @ w_proj     (M=4096, N=512, K=512)
    gemm_bf16_kernel<<<dim3(CC / 128, M / 64), 256, 0, stream>>>(y_bf, wp_t, out, HD, CC);
}
